// DIR_56908316672265
// MI455X (gfx1250) — matrix operand order repaired and checked by reading, not compiled, not run on hardware
//
#include <hip/hip_runtime.h>
#include <hip/hip_bf16.h>

// ---------------------------------------------------------------------------
// Problem constants (match reference)
// ---------------------------------------------------------------------------
static constexpr int N_GRAPHS = 500;
static constexpr int NPG      = 100;     // nodes per graph
static constexpr int EPG      = 800;     // edges per graph
static constexpr int IN_DIM   = 64;
static constexpr int HID      = 128;
static constexpr int NNODES   = N_GRAPHS * NPG;   // 50000
static constexpr int NEDGES   = N_GRAPHS * EPG;   // 400000
static constexpr int KKEEP    = 640;     // int(0.8*800)
static constexpr int KDROP    = EPG - KKEEP;      // 160
static constexpr int NKEEP    = N_GRAPHS * KKEEP; // 320000
static constexpr int NDROP    = N_GRAPHS * KDROP; // 160000

// Output layout (floats, concatenated in return order)
static constexpr size_t OC_X    = 0;
static constexpr size_t OC_EI   = OC_X    + (size_t)NNODES * HID;
static constexpr size_t OC_ATTR = OC_EI   + (size_t)2 * NKEEP;
static constexpr size_t OC_W    = OC_ATTR + (size_t)NKEEP;
static constexpr size_t OC_B    = OC_W    + (size_t)NKEEP;
static constexpr size_t OF_X    = OC_B    + (size_t)NNODES;
static constexpr size_t OF_EI   = OF_X    + (size_t)NNODES * HID;
static constexpr size_t OF_ATTR = OF_EI   + (size_t)2 * NDROP;
static constexpr size_t OF_W    = OF_ATTR + (size_t)NDROP;
static constexpr size_t OF_B    = OF_W    + (size_t)NDROP;
static constexpr size_t OSCORE  = OF_B    + (size_t)NNODES;

#if defined(__has_builtin)
#if __has_builtin(__builtin_amdgcn_global_load_async_to_lds_b128)
#define USE_ASYNC_GATHER 1
#endif
#endif

// ---------------------------------------------------------------------------
// WMMA fragment types
// ---------------------------------------------------------------------------
typedef __attribute__((ext_vector_type(16))) _Float16 v16h;
typedef __attribute__((ext_vector_type(8)))  float    v8f;
typedef __attribute__((ext_vector_type(2)))  _Float16 h2v;
typedef int v4i_vs __attribute__((vector_size(16)));   // async-LDS builtin arg type

// A-fragment (16x32 f16, row-major M x K source), documented VGPR layout:
// lanes 0-15: V0..V3 -> K=0..7, V4..V7 -> K=16..23 (row = lane&15)
// lanes16-31: V0..V3 -> K=8..15, V4..V7 -> K=24..31
__device__ inline v16h load_a_frag_g(const _Float16* __restrict__ A, int K,
                                     int m, int k0, int hi) {
  v16h a;
#pragma unroll
  for (int v = 0; v < 8; ++v) {
    int ka = (v < 4) ? ((hi ? 8 : 0) + 2 * v) : ((hi ? 24 : 16) + 2 * (v - 4));
    h2v p = *reinterpret_cast<const h2v*>(A + (size_t)m * K + k0 + ka);
    a[2 * v] = p[0];
    a[2 * v + 1] = p[1];
  }
  return a;
}

// B-fragment (32x16 f16). Weights pre-transposed: Wt is [N][K], so K-pairs are
// contiguous 32-bit loads.  Same K layout as the A-fragment (V0..V3 -> K=0..7 / 8..15, V4..V7 -> K=16..23 / 24..31).
__device__ inline v16h load_b_frag_g(const _Float16* __restrict__ Wt, int K,
                                     int n, int k0, int hi) {
  v16h b;
#pragma unroll
  for (int v = 0; v < 8; ++v) {
    int kb = (v < 4) ? ((hi ? 8 : 0) + 2 * v) : ((hi ? 24 : 16) + 2 * (v - 4));
    h2v p = *reinterpret_cast<const h2v*>(Wt + (size_t)n * K + k0 + kb);
    b[2 * v] = p[0];
    b[2 * v + 1] = p[1];
  }
  return b;
}

// A-fragment from an LDS tile of row-stride LDK halves.
__device__ inline v16h load_a_frag_lds(const _Float16* rep, int LDK,
                                       int m, int k0, int hi) {
  v16h a;
#pragma unroll
  for (int v = 0; v < 8; ++v) {
    int ka = (v < 4) ? ((hi ? 8 : 0) + 2 * v) : ((hi ? 24 : 16) + 2 * (v - 4));
    h2v p = *reinterpret_cast<const h2v*>(rep + m * LDK + k0 + ka);
    a[2 * v] = p[0];
    a[2 * v + 1] = p[1];
  }
  return a;
}

__device__ inline v8f wmma_f16(v16h a, v16h b, v8f c) {
  return __builtin_amdgcn_wmma_f32_16x16x32_f16(false, a, false, b,
                                                (short)0, c, false, false);
}

// ---------------------------------------------------------------------------
// Conversion / transpose / fill helpers
// ---------------------------------------------------------------------------
__global__ void k_f32_to_f16(const float* __restrict__ s, _Float16* __restrict__ d, int n) {
  int i = blockIdx.x * blockDim.x + threadIdx.x;
  if (i < n) d[i] = (_Float16)s[i];
}

// src is (K, N) row-major f32 ; dst is (N, K) row-major f16
__global__ void k_transpose_f16(const float* __restrict__ s, _Float16* __restrict__ d,
                                int K, int N) {
  int i = blockIdx.x * blockDim.x + threadIdx.x;
  if (i < K * N) {
    int k = i / N, n = i - k * N;
    d[(size_t)n * K + k] = (_Float16)s[i];
  }
}

__global__ void k_fill_f32(float* __restrict__ p, float v, int n) {
  int i = blockIdx.x * blockDim.x + threadIdx.x;
  if (i < n) p[i] = v;
}
__global__ void k_fill_i32(int* __restrict__ p, int v, int n) {
  int i = blockIdx.x * blockDim.x + threadIdx.x;
  if (i < n) p[i] = v;
}

// ---------------------------------------------------------------------------
// WMMA GEMM, register-blocked 16x64 per wave, software-pipelined K loop.
// C[M,N] = A[M,K](f16) * Wt[N,K]^T(f16) + bias.   N % 64 == 0, K % 32 == 0.
// ---------------------------------------------------------------------------
__global__ __launch_bounds__(32)
void k_gemm_wmma(const _Float16* __restrict__ A,
                 const _Float16* __restrict__ Wt,
                 const float* __restrict__ bias,
                 float* __restrict__ C,
                 int M, int N, int K) {
  int lane = threadIdx.x & 31;
  int hi = lane >> 4, row = lane & 15;
  int m0 = blockIdx.x * 16, n0 = blockIdx.y * 64;

  v8f acc[4] = {{}, {}, {}, {}};
  v16h a = load_a_frag_g(A, K, m0 + row, 0, hi);
  v16h b0 = load_b_frag_g(Wt, K, n0 + row, 0, hi);
  v16h b1 = load_b_frag_g(Wt, K, n0 + 16 + row, 0, hi);
  v16h b2 = load_b_frag_g(Wt, K, n0 + 32 + row, 0, hi);
  v16h b3 = load_b_frag_g(Wt, K, n0 + 48 + row, 0, hi);

  for (int k0 = 0; k0 < K; k0 += 32) {
    v16h an = a, bn0 = b0, bn1 = b1, bn2 = b2, bn3 = b3;
    bool more = (k0 + 32) < K;
    if (more) {                      // prefetch next step before the WMMAs
      int kn = k0 + 32;
      an  = load_a_frag_g(A, K, m0 + row, kn, hi);
      bn0 = load_b_frag_g(Wt, K, n0 + row, kn, hi);
      bn1 = load_b_frag_g(Wt, K, n0 + 16 + row, kn, hi);
      bn2 = load_b_frag_g(Wt, K, n0 + 32 + row, kn, hi);
      bn3 = load_b_frag_g(Wt, K, n0 + 48 + row, kn, hi);
    }
    acc[0] = wmma_f16(a, b0, acc[0]);
    acc[1] = wmma_f16(a, b1, acc[1]);
    acc[2] = wmma_f16(a, b2, acc[2]);
    acc[3] = wmma_f16(a, b3, acc[3]);
    a = an; b0 = bn0; b1 = bn1; b2 = bn2; b3 = bn3;
  }

#pragma unroll
  for (int i = 0; i < 4; ++i) {
    int n = n0 + 16 * i + row;
    float bv = bias ? bias[n] : 0.0f;
#pragma unroll
    for (int r = 0; r < 8; ++r) {
      int m = m0 + (hi ? 8 + r : r);
      C[(size_t)m * N + n] = acc[i][r] + bv;
    }
  }
}

// ---------------------------------------------------------------------------
// Per-graph fused message pass + segment-sum in LDS (replaces global atomics):
//   G[d] += (a[src] - b[dst]) * ew   for one graph per block, then optional
//   ReLU and fused f16 conversion on write-back.
// Dynamic LDS: accT(100x128 f32) | bT(100x128 f32) | ew(800) | src(800) | dst(800)
// ---------------------------------------------------------------------------
__global__ __launch_bounds__(256)
void k_graph_messages(const float* __restrict__ a, const float* __restrict__ b,
                      const int* __restrict__ src, const int* __restrict__ dst,
                      const float* __restrict__ ew, float* __restrict__ G,
                      _Float16* __restrict__ h16, int relu) {
  extern __shared__ char smem[];
  float* accT = (float*)smem;             // NPG*HID
  float* bT   = accT + NPG * HID;         // NPG*HID
  float* ewT  = bT + NPG * HID;           // EPG
  int*   srcT = (int*)(ewT + EPG);        // EPG
  int*   dstT = srcT + EPG;               // EPG

  int g = blockIdx.x, t = threadIdx.x;
  int nbase = g * NPG;
  size_t gbase = (size_t)nbase * HID;

  for (int i = t; i < NPG * HID; i += 256) {
    accT[i] = G[gbase + i];               // x@w3 + b3 (pre-accumulated)
    bT[i]   = b[gbase + i];
  }
  for (int i = t; i < EPG; i += 256) {
    int ge = g * EPG + i;
    ewT[i]  = ew[ge];
    srcT[i] = src[ge];
    dstT[i] = dst[ge];
  }
  __syncthreads();

  for (int w = t; w < EPG * HID; w += 256) {
    int e = w >> 7;                       // / HID
    int f = w & (HID - 1);
    int s = srcT[e];
    int dl = dstT[e] - nbase;             // local dst row
    float msg = (a[(size_t)s * HID + f] - bT[dl * HID + f]) * ewT[e];
    atomicAdd(&accT[dl * HID + f], msg);  // ds_add_f32
  }
  __syncthreads();

  for (int i = t; i < NPG * HID; i += 256) {
    float v = accT[i];
    if (relu) v = fmaxf(v, 0.0f);
    G[gbase + i] = v;
    h16[gbase + i] = (_Float16)v;         // fused f16 conversion
  }
}

// ---------------------------------------------------------------------------
// Fused edge MLP: score[e] = relu([h[src]||h[dst]] @ W1 + b1) . w2 + b2
// 16 edges per block; rep tile staged in LDS (async path when available);
// 8 waves x (4 accumulators over a 64-wide N chunk); pipelined B loads.
// ---------------------------------------------------------------------------
__global__ __launch_bounds__(256)
void k_edge_mlp(const _Float16* __restrict__ H16,
                const int* __restrict__ src, const int* __restrict__ dst,
                const _Float16* __restrict__ W1t,  // [512][256] f16
                const float* __restrict__ b1, const float* __restrict__ w2,
                const float* __restrict__ b2, float* __restrict__ scores) {
  __shared__ _Float16 rep[16][2 * HID];   // 16 x 256 f16 = 8KB
  __shared__ float acc_s[16];
  int t = threadIdx.x;
  int e0 = blockIdx.x * 16;

  { // gather [h[src] || h[dst]] rows into LDS: 32 bytes per thread
    int idx = t * 16;
    int m = idx >> 8;                 // edge within tile
    int c = idx & 255;                // column within 256
    int e = e0 + m;
    const _Float16* sr = (c < HID) ? (H16 + (size_t)src[e] * HID + c)
                                   : (H16 + (size_t)dst[e] * HID + (c - HID));
    _Float16* dr = &rep[m][c];
#ifdef USE_ASYNC_GATHER
    __builtin_amdgcn_global_load_async_to_lds_b128(
        (__attribute__((address_space(1))) v4i_vs*)sr,
        (__attribute__((address_space(3))) v4i_vs*)dr, 0, 0);
    __builtin_amdgcn_global_load_async_to_lds_b128(
        (__attribute__((address_space(1))) v4i_vs*)(sr + 8),
        (__attribute__((address_space(3))) v4i_vs*)(dr + 8), 0, 0);
#if defined(__has_builtin) && __has_builtin(__builtin_amdgcn_s_wait_asynccnt)
    __builtin_amdgcn_s_wait_asynccnt(0);
#else
    asm volatile("s_wait_asynccnt 0x0" ::: "memory");
#endif
#else
    const int4* s4 = (const int4*)sr;
    int4* d4 = (int4*)dr;
    d4[0] = s4[0];
    d4[1] = s4[1];
#endif
  }
  if (t < 16) acc_s[t] = 0.0f;

  int lane = t & 31, wave = t >> 5;
  int hi = lane >> 4, row = lane & 15;
  int nb = wave * 64;                 // this wave's 64-wide N chunk

  __builtin_prefetch(W1t + (size_t)(nb + row) * (2 * HID), 0, 3);
  __syncthreads();

  v8f acc[4] = {{}, {}, {}, {}};
  v16h b0 = load_b_frag_g(W1t, 2 * HID, nb + row, 0, hi);
  v16h b1f = load_b_frag_g(W1t, 2 * HID, nb + 16 + row, 0, hi);
  v16h b2f = load_b_frag_g(W1t, 2 * HID, nb + 32 + row, 0, hi);
  v16h b3f = load_b_frag_g(W1t, 2 * HID, nb + 48 + row, 0, hi);

  for (int k0 = 0; k0 < 2 * HID; k0 += 32) {
    v16h a = load_a_frag_lds(&rep[0][0], 2 * HID, row, k0, hi);
    v16h bn0 = b0, bn1 = b1f, bn2 = b2f, bn3 = b3f;
    bool more = (k0 + 32) < 2 * HID;
    if (more) {
      int kn = k0 + 32;
      bn0 = load_b_frag_g(W1t, 2 * HID, nb + row, kn, hi);
      bn1 = load_b_frag_g(W1t, 2 * HID, nb + 16 + row, kn, hi);
      bn2 = load_b_frag_g(W1t, 2 * HID, nb + 32 + row, kn, hi);
      bn3 = load_b_frag_g(W1t, 2 * HID, nb + 48 + row, kn, hi);
    }
    acc[0] = wmma_f16(a, b0, acc[0]);
    acc[1] = wmma_f16(a, b1f, acc[1]);
    acc[2] = wmma_f16(a, b2f, acc[2]);
    acc[3] = wmma_f16(a, b3f, acc[3]);
    b0 = bn0; b1f = bn1; b2f = bn2; b3f = bn3;
  }

#pragma unroll
  for (int i = 0; i < 4; ++i) {
    int n = nb + 16 * i + row;
    float bb = b1[n], ww = w2[n];
#pragma unroll
    for (int r = 0; r < 8; ++r) {
      int m = hi ? 8 + r : r;
      float hv = fmaxf(acc[i][r] + bb, 0.0f);
      atomicAdd(&acc_s[m], hv * ww);    // ds_add_f32
    }
  }
  __syncthreads();
  if (t < 16) scores[e0 + t] = acc_s[t] + b2[0];
}

// ---------------------------------------------------------------------------
// Per-graph top-K: bitonic sort (descending) of 800 scores padded to 1024.
// ---------------------------------------------------------------------------
__global__ __launch_bounds__(512)
void k_sort_topk(const float* __restrict__ scores,
                 int* __restrict__ keepIdx, int* __restrict__ dropIdx) {
  constexpr int SN = 1024;
  __shared__ float key[SN];
  __shared__ int   val[SN];
  int g = blockIdx.x, t = threadIdx.x;
  for (int i = t; i < SN; i += 512) {
    if (i < EPG) { key[i] = scores[g * EPG + i]; val[i] = i; }
    else         { key[i] = -3.0e38f;            val[i] = i; }
  }
  __syncthreads();
  for (int k = 2; k <= SN; k <<= 1) {
    for (int j = k >> 1; j > 0; j >>= 1) {
      for (int i = t; i < SN; i += 512) {
        int ixj = i ^ j;
        if (ixj > i) {
          bool up = ((i & k) != 0);          // final merge: descending
          float ki = key[i], kj = key[ixj];
          bool sw = up ? (ki > kj) : (ki < kj);
          if (sw) {
            key[i] = kj; key[ixj] = ki;
            int tv = val[i]; val[i] = val[ixj]; val[ixj] = tv;
          }
        }
      }
      __syncthreads();
    }
  }
  for (int i = t; i < EPG; i += 512) {
    int gid = g * EPG + val[i];
    if (i < KKEEP) keepIdx[g * KKEEP + i] = gid;
    else           dropIdx[g * KDROP + (i - KKEEP)] = gid;
  }
}

// ---------------------------------------------------------------------------
// Relabel: presence -> exclusive scan -> scatter rows / node-index remap.
// ---------------------------------------------------------------------------
__global__ void k_mark(const int* __restrict__ idxList, int nk,
                       const int* __restrict__ src, const int* __restrict__ dst,
                       int* __restrict__ presence) {
  int j = blockIdx.x * blockDim.x + threadIdx.x;
  if (j < nk) {
    int g = idxList[j];
    presence[src[g]] = 1;
    presence[dst[g]] = 1;
  }
}

__global__ __launch_bounds__(256)
void k_scan1(const int* __restrict__ presence, int* __restrict__ partial,
             int* __restrict__ bsum, int n) {
  __shared__ int s[256];
  int t = threadIdx.x, idx = blockIdx.x * 256 + t;
  int v = (idx < n) ? presence[idx] : 0;
  s[t] = v;
  __syncthreads();
  for (int off = 1; off < 256; off <<= 1) {
    int add = (t >= off) ? s[t - off] : 0;
    __syncthreads();
    s[t] += add;
    __syncthreads();
  }
  if (idx < n) partial[idx] = s[t] - v;   // exclusive within block
  if (t == 255) bsum[blockIdx.x] = s[255];
}

__global__ __launch_bounds__(256)
void k_scan2(int* __restrict__ bsum, int nb) {
  __shared__ int s[256];
  int t = threadIdx.x;
  int v = (t < nb) ? bsum[t] : 0;
  s[t] = v;
  __syncthreads();
  for (int off = 1; off < 256; off <<= 1) {
    int add = (t >= off) ? s[t - off] : 0;
    __syncthreads();
    s[t] += add;
    __syncthreads();
  }
  if (t < nb) bsum[t] = s[t] - v;         // exclusive across blocks
}

__global__ void k_scatter(const int* __restrict__ presence, const int* __restrict__ partial,
                          const int* __restrict__ bsum, const float* __restrict__ h,
                          const int* __restrict__ batch, int* __restrict__ node_idx,
                          float* __restrict__ out_x, float* __restrict__ out_b, int n) {
  int v = blockIdx.x * 256 + threadIdx.x;
  if (v >= n) return;
  if (presence[v]) {
    int pos = bsum[v >> 8] + partial[v];
    node_idx[v] = pos;
    out_b[pos] = (float)batch[v];
    const float4* sr = (const float4*)(h + (size_t)v * HID);
    float4* dr = (float4*)(out_x + (size_t)pos * HID);
#pragma unroll 8
    for (int f = 0; f < HID / 4; ++f) dr[f] = sr[f];
  } else {
    node_idx[v] = -1;
  }
}

__global__ void k_edge_out(const int* __restrict__ idxList, int nk,
                           const int* __restrict__ src, const int* __restrict__ dst,
                           const int* __restrict__ node_idx,
                           const float* __restrict__ eattr, const float* __restrict__ scores,
                           float* __restrict__ out_ei, float* __restrict__ out_attr,
                           float* __restrict__ out_w, float sign) {
  int j = blockIdx.x * blockDim.x + threadIdx.x;
  if (j >= nk) return;
  int g = idxList[j];
  out_ei[j]      = (float)node_idx[src[g]];
  out_ei[nk + j] = (float)node_idx[dst[g]];
  out_attr[j] = eattr[g];
  out_w[j]    = sign * scores[g];
}

__global__ void k_copy_f32(const float* __restrict__ s, float* __restrict__ d, int n) {
  int i = blockIdx.x * blockDim.x + threadIdx.x;
  if (i < n) d[i] = s[i];
}

// ---------------------------------------------------------------------------
// Host side
// ---------------------------------------------------------------------------
static inline int cdiv(int a, int b) { return (a + b - 1) / b; }

extern "C" void kernel_launch(void* const* d_in, const int* in_sizes, int n_in,
                              void* d_out, int out_size, void* d_ws, size_t ws_size,
                              hipStream_t stream) {
  (void)in_sizes; (void)n_in; (void)out_size; (void)ws_size;

  const float* x      = (const float*)d_in[0];
  const int*   ei     = (const int*)d_in[1];
  const float* eattr  = (const float*)d_in[2];
  const int*   batch  = (const int*)d_in[3];
  const float* c1w1   = (const float*)d_in[4];
  const float* c1b1   = (const float*)d_in[5];
  const float* c1w2   = (const float*)d_in[6];
  const float* c1w3   = (const float*)d_in[7];
  const float* c1b3   = (const float*)d_in[8];
  const float* c2w1   = (const float*)d_in[9];
  const float* c2b1   = (const float*)d_in[10];
  const float* c2w2   = (const float*)d_in[11];
  const float* c2w3   = (const float*)d_in[12];
  const float* c2b3   = (const float*)d_in[13];
  const float* mlpw1  = (const float*)d_in[14];
  const float* mlpb1  = (const float*)d_in[15];
  const float* mlpw2  = (const float*)d_in[16];
  const float* mlpb2  = (const float*)d_in[17];

  const int* srcA = ei;
  const int* dstA = ei + NEDGES;

  // ---- workspace layout (bytes) ----
  char* ws = (char*)d_ws;
  float*    A    = (float*)(ws + 0);                         // 50000x128 f32
  float*    B    = (float*)(ws + 25600000);                  // 50000x128 f32
  float*    G    = (float*)(ws + 51200000);                  // 50000x128 f32 (h)
  _Float16* H16  = (_Float16*)(ws + 76800000);               // 50000x128 f16
  _Float16* W16  = (_Float16*)(ws + 89600000);               // transposed weights
  _Float16* w1t1 = W16 + 0;            // 128x64
  _Float16* w2t1 = W16 + 8192;         // 128x64
  _Float16* w3t1 = W16 + 16384;        // 128x64
  _Float16* w1t2 = W16 + 24576;        // 128x128
  _Float16* w2t2 = W16 + 40960;        // 128x128
  _Float16* w3t2 = W16 + 57344;        // 128x128
  _Float16* mw1t = W16 + 73728;        // 512x256
  float* scores  = (float*)(ws + 90009600);                  // 400000 f32
  int*   keepIdx = (int*)(ws + 91609600);                    // 320000
  int*   dropIdx = (int*)(ws + 92889600);                    // 160000
  int*   pres    = (int*)(ws + 93529600);                    // 50000
  int*   part    = (int*)(ws + 93729600);                    // 50000
  int*   bsum    = (int*)(ws + 93929600);                    // 256
  int*   nidx    = (int*)(ws + 93930624);                    // 50000

  float* out = (float*)d_out;
  const int NB = cdiv(NNODES, 256);    // scan blocks (196)
  const int MSG_LDS = (2 * NPG * HID + 3 * EPG) * 4;   // 112000 bytes

  // ---- weight conversion (transposed f16) ----
  k_transpose_f16<<<cdiv(IN_DIM * HID, 256), 256, 0, stream>>>(c1w1, w1t1, IN_DIM, HID);
  k_transpose_f16<<<cdiv(IN_DIM * HID, 256), 256, 0, stream>>>(c1w2, w2t1, IN_DIM, HID);
  k_transpose_f16<<<cdiv(IN_DIM * HID, 256), 256, 0, stream>>>(c1w3, w3t1, IN_DIM, HID);
  k_transpose_f16<<<cdiv(HID * HID, 256), 256, 0, stream>>>(c2w1, w1t2, HID, HID);
  k_transpose_f16<<<cdiv(HID * HID, 256), 256, 0, stream>>>(c2w2, w2t2, HID, HID);
  k_transpose_f16<<<cdiv(HID * HID, 256), 256, 0, stream>>>(c2w3, w3t2, HID, HID);
  k_transpose_f16<<<cdiv(2 * HID * 4 * HID, 256), 256, 0, stream>>>(mlpw1, mw1t, 2 * HID, 4 * HID);
  k_f32_to_f16<<<cdiv(NNODES * IN_DIM, 256), 256, 0, stream>>>(x, H16, NNODES * IN_DIM);

  dim3 gemmGrid(NNODES / 16, HID / 64);   // 16x64 tiles per wave

  // ---- LEConv layer 1 (A input f16 = x, K=64) ----
  k_gemm_wmma<<<gemmGrid, 32, 0, stream>>>(H16, w1t1, c1b1, A, NNODES, HID, IN_DIM);
  k_gemm_wmma<<<gemmGrid, 32, 0, stream>>>(H16, w2t1, nullptr, B, NNODES, HID, IN_DIM);
  k_gemm_wmma<<<gemmGrid, 32, 0, stream>>>(H16, w3t1, c1b3, G, NNODES, HID, IN_DIM);
  k_graph_messages<<<N_GRAPHS, 256, MSG_LDS, stream>>>(A, B, srcA, dstA, eattr, G, H16, 1);

  // ---- LEConv layer 2 (A input f16 = h1, K=128) ----
  k_gemm_wmma<<<gemmGrid, 32, 0, stream>>>(H16, w1t2, c2b1, A, NNODES, HID, HID);
  k_gemm_wmma<<<gemmGrid, 32, 0, stream>>>(H16, w2t2, nullptr, B, NNODES, HID, HID);
  k_gemm_wmma<<<gemmGrid, 32, 0, stream>>>(H16, w3t2, c2b3, G, NNODES, HID, HID);
  k_graph_messages<<<N_GRAPHS, 256, MSG_LDS, stream>>>(A, B, srcA, dstA, eattr, G, H16, 0);

  // ---- fused edge-scoring MLP (dominant WMMA work) ----
  k_edge_mlp<<<NEDGES / 16, 256, 0, stream>>>(H16, srcA, dstA, mw1t, mlpb1, mlpw2, mlpb2, scores);
  k_copy_f32<<<cdiv(NEDGES, 256), 256, 0, stream>>>(scores, out + OSCORE, NEDGES);

  // ---- per-graph top-K ----
  k_sort_topk<<<N_GRAPHS, 512, 0, stream>>>(scores, keepIdx, dropIdx);

  // ---- fill pooled outputs (rows beyond the unique count) ----
  k_fill_f32<<<cdiv(NNODES * HID, 256), 256, 0, stream>>>(out + OC_X, 0.0f, NNODES * HID);
  k_fill_f32<<<cdiv(NNODES, 256), 256, 0, stream>>>(out + OC_B, -1.0f, NNODES);
  k_fill_f32<<<cdiv(NNODES * HID, 256), 256, 0, stream>>>(out + OF_X, 0.0f, NNODES * HID);
  k_fill_f32<<<cdiv(NNODES, 256), 256, 0, stream>>>(out + OF_B, -1.0f, NNODES);

  // ---- relabel for kept edges ----
  k_fill_i32<<<cdiv(NNODES, 256), 256, 0, stream>>>(pres, 0, NNODES);
  k_mark<<<cdiv(NKEEP, 256), 256, 0, stream>>>(keepIdx, NKEEP, srcA, dstA, pres);
  k_scan1<<<NB, 256, 0, stream>>>(pres, part, bsum, NNODES);
  k_scan2<<<1, 256, 0, stream>>>(bsum, NB);
  k_scatter<<<NB, 256, 0, stream>>>(pres, part, bsum, G, batch, nidx,
                                    out + OC_X, out + OC_B, NNODES);
  k_edge_out<<<cdiv(NKEEP, 256), 256, 0, stream>>>(keepIdx, NKEEP, srcA, dstA, nidx,
                                                   eattr, scores,
                                                   out + OC_EI, out + OC_ATTR, out + OC_W, 1.0f);

  // ---- relabel for dropped edges ----
  k_fill_i32<<<cdiv(NNODES, 256), 256, 0, stream>>>(pres, 0, NNODES);
  k_mark<<<cdiv(NDROP, 256), 256, 0, stream>>>(dropIdx, NDROP, srcA, dstA, pres);
  k_scan1<<<NB, 256, 0, stream>>>(pres, part, bsum, NNODES);
  k_scan2<<<1, 256, 0, stream>>>(bsum, NB);
  k_scatter<<<NB, 256, 0, stream>>>(pres, part, bsum, G, batch, nidx,
                                    out + OF_X, out + OF_B, NNODES);
  k_edge_out<<<cdiv(NDROP, 256), 256, 0, stream>>>(dropIdx, NDROP, srcA, dstA, nidx,
                                                   eattr, scores,
                                                   out + OF_EI, out + OF_ATTR, out + OF_W, -1.0f);
}
